// MultiHeadAttention_29394756174088
// MI455X (gfx1250) — matrix operand order repaired and checked by reading, not compiled, not run on hardware
//
#include <hip/hip_runtime.h>
#include <hip/hip_bf16.h>

// Problem constants (from reference): B=2, L=2048, D=1024, H=16, HD=64
#define BB 2
#define LL 2048
#define DD 1024
#define HH 16
#define HD 64
#define NEGBIG (-1.0e9f)

typedef __attribute__((ext_vector_type(16))) __bf16 v16bf;
typedef __attribute__((ext_vector_type(8)))  __bf16 v8bf;
typedef __attribute__((ext_vector_type(8)))  float  v8f;

// ---------------------------------------------------------------------------
// helpers
// ---------------------------------------------------------------------------
static __device__ inline v8f wmma_bf16(v16bf a, v16bf b, v8f c) {
    // D = A(16x32 bf16) * B(32x16 bf16) + C(16x16 f32)
    return __builtin_amdgcn_wmma_f32_16x16x32_bf16(
        /*neg_a=*/false, a, /*neg_b=*/false, b,
        /*c_mod=*/(short)0, c, /*reuse_a=*/false, /*reuse_b=*/false);
}

static __device__ inline v16bf cat8(v8bf lo, v8bf hi) {
    v16bf r;
#pragma unroll
    for (int i = 0; i < 8; ++i) { r[i] = lo[i]; r[i + 8] = hi[i]; }
    return r;
}

// Convert 4x float4 (16 contiguous f32) -> v16bf
static __device__ inline v16bf cvt16f4(float4 x0, float4 x1, float4 x2, float4 x3) {
    v16bf r;
    r[0]=(__bf16)x0.x; r[1]=(__bf16)x0.y; r[2]=(__bf16)x0.z; r[3]=(__bf16)x0.w;
    r[4]=(__bf16)x1.x; r[5]=(__bf16)x1.y; r[6]=(__bf16)x1.z; r[7]=(__bf16)x1.w;
    r[8]=(__bf16)x2.x; r[9]=(__bf16)x2.y; r[10]=(__bf16)x2.z; r[11]=(__bf16)x2.w;
    r[12]=(__bf16)x3.x; r[13]=(__bf16)x3.y; r[14]=(__bf16)x3.z; r[15]=(__bf16)x3.w;
    return r;
}

static __device__ inline v16bf cvt16f(const float* p) {
    const float4* q = (const float4*)p;
    return cvt16f4(q[0], q[1], q[2], q[3]);
}

// Async global -> LDS copy of one 16B chunk (CDNA5 async path, ASYNCcnt)
static __device__ inline void async_copy_b128(unsigned lds_off, const void* gptr) {
    asm volatile("global_load_async_to_lds_b128 %0, %1, off"
                 :: "v"(lds_off), "v"(gptr) : "memory");
}
static __device__ inline void wait_async0() {
    asm volatile("s_wait_asynccnt 0x0" ::: "memory");
}

// ---------------------------------------------------------------------------
// GEMM: C[m,n] = sum_k A[m,k] * W[n,k] + bias[n]
//   M = B*L = 4096, N = K = D = 1024
//   mode 0: out bf16, head-split      [B,H,L,HD]  (Q, K projections)
//   mode 1: out bf16, head-transposed [B,H,HD,L]  (V projection)
//   mode 2: out f32, plain [M,N]                  (final output projection)
// Block: 256 threads = 8 waves; wave w owns 16x16 tile at column w*16.
// Grid: (N/128, M/16). A tile staged via global_load_async_to_lds_b128.
// ---------------------------------------------------------------------------
template <bool ABF16>
__global__ __launch_bounds__(256)
void gemm_wt(const void* __restrict__ A_, const float* __restrict__ W,
             const float* __restrict__ bias,
             __bf16* __restrict__ outb, float* __restrict__ outf, int mode) {
    // raw A tile: 16 rows x 32 k, f32 (2KB) or bf16 (1KB)
    __shared__ __align__(16) unsigned char Asraw[16 * 32 * 4];

    const int tid  = threadIdx.x;
    const int w    = tid >> 5;
    const int lane = tid & 31;
    const int lrow = lane & 15;
    const int sel  = (lane >> 4) & 1;      // 0: lanes 0-15, 1: lanes 16-31
    const int half = sel ? 8 : 0;          // C-fragment row offset

    const int mbase = blockIdx.y * 16;
    const int nbase = blockIdx.x * 128 + w * 16;
    const int nglob = nbase + lrow;

    const int ESZ      = ABF16 ? 2 : 4;        // element bytes
    const int ROWB     = 32 * ESZ;             // tile row bytes
    const int NCHUNK   = (16 * ROWB) / 16;     // 16B chunks per tile

    v8f acc;
#pragma unroll
    for (int i = 0; i < 8; ++i) acc[i] = 0.0f;

    for (int k0 = 0; k0 < DD; k0 += 32) {
        __syncthreads();
        // async stage of A tile (memory -> LDS, no VGPR round trip)
        for (int e = tid; e < NCHUNK; e += 256) {
            int r  = (e * 16) / ROWB;            // tile row
            int cb = (e * 16) % ROWB;            // byte offset within row
            const unsigned char* gp = (const unsigned char*)A_
                + ((size_t)(mbase + r) * DD + k0) * ESZ + cb;
            unsigned lds_off = (unsigned)(size_t)&Asraw[e * 16];
            async_copy_b128(lds_off, gp);
        }
        wait_async0();
        __syncthreads();

        // A fragment: row = lrow; lanes<16 k-window {0..7,16..23}, lanes>=16 {8..15,24..31}
        v16bf afrag;
        if (ABF16) {
            const v8bf* asv = (const v8bf*)Asraw;          // 4 chunks of 8 per row
            afrag = cat8(asv[lrow * 4 + sel], asv[lrow * 4 + 2 + sel]);
        } else {
            const float4* f4 = (const float4*)Asraw;       // 8 float4 per row
            float4 p0 = f4[lrow * 8 + sel * 2 + 0];
            float4 p1 = f4[lrow * 8 + sel * 2 + 1];
            float4 p2 = f4[lrow * 8 + 4 + sel * 2 + 0];
            float4 p3 = f4[lrow * 8 + 4 + sel * 2 + 1];
            afrag = cvt16f4(p0, p1, p2, p3);
        }

        // B fragment: column n = nglob; lanes<16 k {0..7,16..23}, lanes>=16 {8..15,24..31}
        const float4* wq = (const float4*)(W + (size_t)nglob * DD + k0 + sel * 8);
        v16bf bfrag = cvt16f4(wq[0], wq[1], wq[4], wq[5]);

        acc = wmma_bf16(afrag, bfrag, acc);
    }

    const float bval = bias[nglob];
#pragma unroll
    for (int r = 0; r < 8; ++r) {
        float v = acc[r] + bval;
        int m = mbase + r + half;
        if (mode == 2) {
            outf[(size_t)m * DD + nglob] = v;
        } else {
            int bb = m >> 11, l = m & (LL - 1);      // L = 2048
            int hh = nglob >> 6, hd = nglob & 63;    // HD = 64
            size_t dst = (mode == 0)
                ? ((((size_t)bb * HH + hh) * LL + l) * HD + hd)
                : ((((size_t)bb * HH + hh) * HD + hd) * LL + l);
            outb[dst] = (__bf16)v;
        }
    }
}

// ---------------------------------------------------------------------------
// Fused attention: per block = (b, h, 16-query tile); 4 waves x 512 keys each.
// Pass 1: online per-lane (max, sumexp) over QK^T tiles (recompute-flash).
// Combine stats via LDS. Pass 2: recompute scores, write exact softmax probs
// to global attn (only traversal of the 512MB tensor), transpose P tile via
// per-wave LDS, accumulate P @ V with bf16 WMMA. Cross-wave O reduce via LDS.
// ---------------------------------------------------------------------------
__global__ __launch_bounds__(128)
void attn_kernel(const __bf16* __restrict__ Qb, const __bf16* __restrict__ Kb,
                 const __bf16* __restrict__ Vb, const int* __restrict__ mask,
                 float* __restrict__ attn, __bf16* __restrict__ Ctx) {
    __shared__ float  pm[16][64];
    __shared__ float  pl[16][64];
    __shared__ float  rowM[16];
    __shared__ float  rowLinv[16];
    __shared__ __bf16 Pt[4][16][32];       // per-wave P-tile transpose buffer
    __shared__ float  Osum[4][16][64];     // per-wave partial outputs

    const int qt = blockIdx.x, h = blockIdx.y, b = blockIdx.z;
    const int bh = b * HH + h;
    const int qbase = qt * 16;
    const int tid  = threadIdx.x;
    const int w    = tid >> 5;
    const int lane = tid & 31;
    const int lrow = lane & 15;
    const int sel  = (lane >> 4) & 1;
    const int half = sel ? 8 : 0;
    const float scale = 0.125f;            // 1/sqrt(64)
    const size_t maskbase = (size_t)b * LL * LL;

    // Q fragments (loop invariant): row = query qbase+lrow, 64-d row = 8 v8bf chunks
    const v8bf* qv = (const v8bf*)(Qb + ((size_t)bh * LL + qbase + lrow) * HD);
    const v16bf a0 = cat8(qv[sel * 2 + 0], qv[sel * 2 + 1]);   // d 0..31 window
    const v16bf a1 = cat8(qv[4 + sel * 2 + 0], qv[4 + sel * 2 + 1]);   // d 32..63 window

    const int kw0 = w * 512;               // this wave's key range [kw0, kw0+512)

    // ---------------- pass 1: online row stats (per-lane, no shuffles) ------
    float m8[8], l8[8];
#pragma unroll
    for (int r = 0; r < 8; ++r) { m8[r] = -3.0e38f; l8[r] = 0.0f; }

    for (int kt = 0; kt < 32; ++kt) {
        const int kk = kw0 + kt * 16;
        const v8bf* kv = (const v8bf*)(Kb + ((size_t)bh * LL + kk + lrow) * HD);
        // prefetch next key tile row into caches (global_prefetch_b8)
        if (kt + 1 < 32)
            __builtin_prefetch(Kb + ((size_t)bh * LL + kk + 16 + lrow) * HD, 0, 3);
        v16bf b0 = cat8(kv[sel * 2 + 0], kv[sel * 2 + 1]);         // d 0..31
        v16bf b1 = cat8(kv[4 + sel * 2 + 0], kv[4 + sel * 2 + 1]); // d 32..63
        v8f c;
#pragma unroll
        for (int i = 0; i < 8; ++i) c[i] = 0.0f;
        c = wmma_bf16(a0, b0, c);
        c = wmma_bf16(a1, b1, c);
#pragma unroll
        for (int r = 0; r < 8; ++r) {
            int row = r + half;
            int mv = mask[maskbase + (size_t)(qbase + row) * LL + kk + lrow];
            float s = (mv == 0) ? NEGBIG : c[r] * scale;
            float mn = fmaxf(m8[r], s);
            l8[r] = l8[r] * __expf(m8[r] - mn) + __expf(s - mn);
            m8[r] = mn;
        }
    }
#pragma unroll
    for (int r = 0; r < 8; ++r) {
        int row = r + half;
        pm[row][w * 16 + lrow] = m8[r];
        pl[row][w * 16 + lrow] = l8[r];
    }
    __syncthreads();
    if (tid < 16) {
        float M = -3.0e38f;
        for (int j = 0; j < 64; ++j) M = fmaxf(M, pm[tid][j]);
        float Ls = 0.0f;
        for (int j = 0; j < 64; ++j) Ls += pl[tid][j] * __expf(pm[tid][j] - M);
        rowM[tid] = M;
        rowLinv[tid] = 1.0f / Ls;
    }
    __syncthreads();

    // ---------------- pass 2: probs -> global attn, P @ V accumulation -----
    v8f oacc[4];
#pragma unroll
    for (int dc = 0; dc < 4; ++dc)
#pragma unroll
        for (int i = 0; i < 8; ++i) oacc[dc][i] = 0.0f;

    for (int pr = 0; pr < 16; ++pr) {
        const int pbase = kw0 + pr * 32;
        // prefetch V rows for this pair (one per d-chunk; rows are 4KB apart)
#pragma unroll
        for (int dc = 0; dc < 4; ++dc)
            __builtin_prefetch(Vb + ((size_t)bh * HD + dc * 16 + lrow) * LL + pbase, 0, 3);
#pragma unroll
        for (int sub = 0; sub < 2; ++sub) {
            const int kk = pbase + sub * 16;
            const v8bf* kv = (const v8bf*)(Kb + ((size_t)bh * LL + kk + lrow) * HD);
            v16bf b0 = cat8(kv[sel * 2 + 0], kv[sel * 2 + 1]);
            v16bf b1 = cat8(kv[4 + sel * 2 + 0], kv[4 + sel * 2 + 1]);
            v8f c;
#pragma unroll
            for (int i = 0; i < 8; ++i) c[i] = 0.0f;
            c = wmma_bf16(a0, b0, c);
            c = wmma_bf16(a1, b1, c);
#pragma unroll
            for (int r = 0; r < 8; ++r) {
                int row = r + half;
                int mv = mask[maskbase + (size_t)(qbase + row) * LL + kk + lrow];
                float s = (mv == 0) ? NEGBIG : c[r] * scale;
                float p = __expf(s - rowM[row]) * rowLinv[row];
                attn[((size_t)bh * LL + qbase + row) * LL + kk + lrow] = p;
                Pt[w][row][sub * 16 + lrow] = (__bf16)p;
            }
        }
        // A fragment from transposed P tile (same-wave LDS; DS ops are in-order)
        const v8bf* ptv = (const v8bf*)Pt[w][lrow];
        v16bf ap = cat8(ptv[sel * 2], ptv[sel * 2 + 1]);
        // B fragments straight from head-transposed V: contiguous 16 bf16 per lane
#pragma unroll
        for (int dc = 0; dc < 4; ++dc) {
            const v8bf* vp = (const v8bf*)(Vb + ((size_t)bh * HD + dc * 16 + lrow) * LL
                                              + pbase + sel * 16);
            v16bf bv = cat8(vp[0], vp[1]);
            oacc[dc] = wmma_bf16(ap, bv, oacc[dc]);
        }
    }

    // ---------------- cross-wave output reduction -> Ctx (bf16) ------------
#pragma unroll
    for (int dc = 0; dc < 4; ++dc)
#pragma unroll
        for (int r = 0; r < 8; ++r)
            Osum[w][r + half][dc * 16 + lrow] = oacc[dc][r];
    __syncthreads();
    {
        int idx = tid * 8;                   // 128 threads x 8 = 16*64 elems
        int row = idx >> 6, d0 = idx & 63;
        v8bf ov;
#pragma unroll
        for (int i = 0; i < 8; ++i) {
            float s = Osum[0][row][d0 + i] + Osum[1][row][d0 + i]
                    + Osum[2][row][d0 + i] + Osum[3][row][d0 + i];
            ov[i] = (__bf16)s;
        }
        *(v8bf*)(Ctx + ((size_t)(b * LL + qbase + row)) * DD + h * HD + d0) = ov;
    }
}

// ---------------------------------------------------------------------------
// launch: inputs are q,k,v,mask,Wq,bq,Wk,bk,Wv,bv,Wo,bo
// d_out = [out (B*L*D f32)] ++ [attn (B*H*L*L f32)]
// workspace: Qb,Kb,Vb,Ctx bf16 = 4 x 8MB = 32MB
// ---------------------------------------------------------------------------
extern "C" void kernel_launch(void* const* d_in, const int* in_sizes, int n_in,
                              void* d_out, int out_size, void* d_ws, size_t ws_size,
                              hipStream_t stream) {
    const float* q    = (const float*)d_in[0];
    const float* k    = (const float*)d_in[1];
    const float* v    = (const float*)d_in[2];
    const int*   mask = (const int*)d_in[3];
    const float* Wq = (const float*)d_in[4];  const float* bq = (const float*)d_in[5];
    const float* Wk = (const float*)d_in[6];  const float* bk = (const float*)d_in[7];
    const float* Wv = (const float*)d_in[8];  const float* bv = (const float*)d_in[9];
    const float* Wo = (const float*)d_in[10]; const float* bo = (const float*)d_in[11];

    const size_t NTOK = (size_t)BB * LL * DD;       // 4194304
    __bf16* Qb  = (__bf16*)d_ws;
    __bf16* Kb  = Qb + NTOK;
    __bf16* Vb  = Kb + NTOK;
    __bf16* Ctx = Vb + NTOK;

    float* out  = (float*)d_out;
    float* attn = out + NTOK;

    dim3 gblk(256), ggrid(DD / 128, (BB * LL) / 16);   // (8, 256)
    gemm_wt<false><<<ggrid, gblk, 0, stream>>>(q, Wq, bq, Qb, nullptr, 0);
    gemm_wt<false><<<ggrid, gblk, 0, stream>>>(k, Wk, bk, Kb, nullptr, 0);
    gemm_wt<false><<<ggrid, gblk, 0, stream>>>(v, Wv, bv, Vb, nullptr, 1);

    dim3 agrid(LL / 16, HH, BB);                       // (128, 16, 2)
    attn_kernel<<<agrid, dim3(128), 0, stream>>>(Qb, Kb, Vb, mask, attn, Ctx);

    gemm_wt<true><<<ggrid, gblk, 0, stream>>>(Ctx, Wo, bo, nullptr, out, 2);
}
